// GNN_37615323579234
// MI455X (gfx1250) — matrix operand order repaired and checked by reading, not compiled, not run on hardware
//
#include <hip/hip_runtime.h>
#include <hip/hip_bf16.h>

// ---------------------------------------------------------------------------
// GNN forward for MI455X (gfx1250): bf16 WMMA GEMMs for the A@H products,
// double-buffered async global->LDS staging for shared B tiles, fp32 accum.
// If workspace allows, A is pre-converted to bf16 once (fits the 192MB L2).
// ---------------------------------------------------------------------------

typedef __attribute__((ext_vector_type(16))) __bf16 v16bf;
typedef __attribute__((ext_vector_type(8)))  float  v8f;
typedef __attribute__((ext_vector_type(4)))  float  v4f;
typedef int v4i __attribute__((vector_size(16)));
typedef __attribute__((address_space(1))) v4i gas_v4i;   // global AS vector
typedef __attribute__((address_space(3))) v4i las_v4i;   // LDS AS vector

#ifdef __has_builtin
#if __has_builtin(__builtin_amdgcn_global_load_async_to_lds_b128) && \
    __has_builtin(__builtin_amdgcn_s_wait_asynccnt)
#define USE_ASYNC_LDS 1
#endif
#endif
#ifndef USE_ASYNC_LDS
#define USE_ASYNC_LDS 0
#endif

// --- helpers ---------------------------------------------------------------

__device__ __forceinline__ unsigned packbf(float x, float y) {
  // round-to-nearest-even fp32 -> bf16, packed pair (x low, y high)
  unsigned ux = __float_as_uint(x);
  unsigned uy = __float_as_uint(y);
  ux += 0x7FFFu + ((ux >> 16) & 1u);
  uy += 0x7FFFu + ((uy >> 16) & 1u);
  return (ux >> 16) | (uy & 0xFFFF0000u);
}

__device__ __forceinline__ __bf16 f2bf(float f) {
  unsigned u = __float_as_uint(f);
  u += 0x7FFFu + ((u >> 16) & 1u);
  unsigned short s = (unsigned short)(u >> 16);
  return __builtin_bit_cast(__bf16, s);
}

__device__ __forceinline__ float bf2f(__bf16 b) {
  unsigned short s = __builtin_bit_cast(unsigned short, b);
  return __uint_as_float(((unsigned)s) << 16);
}

// 16-byte global -> LDS copy, async on CDNA5 when the builtin exists.
__device__ __forceinline__ void copy16_g2l(void* lds_dst, const void* g_src) {
#if USE_ASYNC_LDS
  __builtin_amdgcn_global_load_async_to_lds_b128(
      (gas_v4i*)(unsigned long long)g_src,
      (las_v4i*)(unsigned)(unsigned long long)lds_dst,
      0, 0);
#else
  *(uint4*)lds_dst = *(const uint4*)g_src;
#endif
}

__device__ __forceinline__ void wait_async_all() {
#if USE_ASYNC_LDS
  __builtin_amdgcn_s_wait_asynccnt(0);
#endif
}

__device__ __forceinline__ v8f wmma_bf16(v16bf a, v16bf b, v8f c) {
  return __builtin_amdgcn_wmma_f32_16x16x32_bf16(
      /*neg_a=*/false, a, /*neg_b=*/false, b,
      /*c_mod=*/(short)0, c, /*reuse_a=*/false, /*reuse_b=*/false);
}

union AFrag { unsigned u[8]; uint4 q[2]; v16bf v; };
union BFrag { uint4 q[2]; v16bf v; };

// --- fp32 -> bf16 bulk conversion (A matrices), NT loads (read-once) -------

__global__ __launch_bounds__(256) void convert_kernel(
    const float* __restrict__ A, __bf16* __restrict__ Ab, size_t count8) {
  const size_t i = (size_t)blockIdx.x * 256 + threadIdx.x;
  if (i >= count8) return;
  const v4f* p = (const v4f*)(A + i * 8);
  const v4f x = __builtin_nontemporal_load(p);
  const v4f y = __builtin_nontemporal_load(p + 1);
  uint4 o;
  o.x = packbf(x.x, x.y); o.y = packbf(x.z, x.w);
  o.z = packbf(y.x, y.y); o.w = packbf(y.z, y.w);
  *(uint4*)(Ab + i * 8) = o;
}

// --- dual-output GEMM: Y = A @ H, H given transposed bf16 [64 x N] ---------
// 128-thread WG (4 waves), 64 output rows per WG, full 64-col output width.
// K tiled by 64; B tiles double-buffered in LDS via async loads.
// DUAL: compute two outputs sharing the A stream (A_ue@H_ue and A_ue@H_cl).
// ABF:  1 = A is pre-converted bf16 (direct fragment loads),
//       0 = A is fp32, converted to bf16 in registers on the fly.

template <int DUAL, int ABF>
__global__ __launch_bounds__(128) void gemm_kernel(
    const void* __restrict__ Araw,     // [n x n] row-major (fp32 or bf16)
    const __bf16* __restrict__ Ht1,    // [64 x n] (H1 transposed)
    const __bf16* __restrict__ Ht2,    // [64 x n] (only if DUAL)
    float* __restrict__ Y1,            // [n x 64]
    float* __restrict__ Y2,            // [n x 64] (only if DUAL)
    int n) {
  constexpr int LDK  = 64 + 8;         // padded bf16 K-pitch (bank-friendly)
  constexpr int TILE = 64 * LDK;       // bf16 elems per staged tile
  __shared__ __bf16 lds1[2 * TILE];
  __shared__ __bf16 lds2[DUAL ? 2 * TILE : 1];

  const int tid  = threadIdx.x;
  const int lane = tid & 31;
  const int wave = tid >> 5;                       // 0..3
  const int rowBase = blockIdx.x * 64 + wave * 16; // 16-row strip per wave
  const int mrow  = rowBase + (lane & 15);
  const int hi    = lane >> 4;                     // 0 / 1 (lane half)
  const int koffA = hi * 8;                        // A frag K sub-offset

  v8f acc1[4], acc2[4];
#pragma unroll
  for (int c = 0; c < 4; ++c)
#pragma unroll
    for (int i = 0; i < 8; ++i) { acc1[c][i] = 0.f; acc2[c][i] = 0.f; }

  const int colLd = tid >> 1;          // 0..63 (staging column)
  const int subLd = (tid & 1) * 8;     // bf16 element offset

  // stage one 64x64-K bf16 tile (8 KB) per B matrix into LDS buffer `buf`
  auto stage = [&](int buf, int k0) {
#pragma unroll
    for (int r = 0; r < 4; ++r) {
      const int ke = subLd + r * 16;
      copy16_g2l(&lds1[buf * TILE + colLd * LDK + ke],
                 Ht1 + (size_t)colLd * n + k0 + ke);
    }
    if constexpr (DUAL) {
#pragma unroll
      for (int r = 0; r < 4; ++r) {
        const int ke = subLd + r * 16;
        copy16_g2l(&lds2[buf * TILE + colLd * LDK + ke],
                   Ht2 + (size_t)colLd * n + k0 + ke);
      }
    }
  };

  // prologue: fill buffer 0
  stage(0, 0);
  wait_async_all();
  __syncthreads();

  for (int k0 = 0; k0 < n; k0 += 64) {
    const int buf   = (k0 >> 6) & 1;
    const int knext = k0 + 64;
    if (knext < n) stage(buf ^ 1, knext);          // async: overlap with WMMA

    // ---- A fragments for both 32-K substeps -----------------------------
    AFrag af[2];
    if constexpr (ABF) {
      // bf16 A: fragments load directly in WMMA register layout
      const __bf16* arow =
          (const __bf16*)Araw + (size_t)mrow * n + k0 + koffA;
      if (knext < n) __builtin_prefetch(arow + 64, 0, 0);
      af[0].q[0] = *(const uint4*)(arow + 0);      // K koffA..+7
      af[0].q[1] = *(const uint4*)(arow + 16);     // K koffA+16..+23
      af[1].q[0] = *(const uint4*)(arow + 32);
      af[1].q[1] = *(const uint4*)(arow + 48);
    } else {
      // fp32 A: load and convert in registers
      const float* arow = (const float*)Araw + (size_t)mrow * n + k0;
      if (knext < n) __builtin_prefetch(arow + 128, 0, 0);
      float4 a0[4], a1[4];
#pragma unroll
      for (int q = 0; q < 4; ++q) {
        // K offsets {0,4,16,20} within each 32-K substep
        const int off = ((q >> 1) * 16) + ((q & 1) * 4) + koffA;
        a0[q] = *(const float4*)(arow + off);
        a1[q] = *(const float4*)(arow + 32 + off);
      }
#pragma unroll
      for (int q = 0; q < 4; ++q) {
        af[0].u[2 * q + 0] = packbf(a0[q].x, a0[q].y);
        af[0].u[2 * q + 1] = packbf(a0[q].z, a0[q].w);
        af[1].u[2 * q + 0] = packbf(a1[q].x, a1[q].y);
        af[1].u[2 * q + 1] = packbf(a1[q].z, a1[q].w);
      }
    }

#pragma unroll
    for (int s = 0; s < 2; ++s) {                  // two K=32 WMMA substeps
      // batch all B-frag LDS loads (one ds clause, one wait), then the WMMAs
      BFrag b1[4], b2[DUAL ? 4 : 1];
#pragma unroll
      for (int c = 0; c < 4; ++c) {
        const int boff =
            buf * TILE + (c * 16 + (lane & 15)) * LDK + s * 32 + hi * 8;
        b1[c].q[0] = *(const uint4*)&lds1[boff];
        b1[c].q[1] = *(const uint4*)&lds1[boff + 16];
        if constexpr (DUAL) {
          b2[c].q[0] = *(const uint4*)&lds2[boff];
          b2[c].q[1] = *(const uint4*)&lds2[boff + 16];
        }
      }
#pragma unroll
      for (int c = 0; c < 4; ++c) {
        acc1[c] = wmma_bf16(af[s].v, b1[c].v, acc1[c]);
        if constexpr (DUAL) acc2[c] = wmma_bf16(af[s].v, b2[c].v, acc2[c]);
      }
    }

    wait_async_all();                              // next tile landed in LDS
    __syncthreads();                               // everyone done with `buf`
  }

  // ---- epilogue: D layout = vgpr v -> row v (lanes 0-15) / v+8 (16-31) ----
  const int col0  = lane & 15;
  const int rbase = rowBase + hi * 8;
#pragma unroll
  for (int c = 0; c < 4; ++c) {
    const int col = c * 16 + col0;
#pragma unroll
    for (int v = 0; v < 8; ++v) {
      Y1[(size_t)(rbase + v) * 64 + col] = acc1[c][v];
      if constexpr (DUAL) Y2[(size_t)(rbase + v) * 64 + col] = acc2[c][v];
    }
  }
}

// --- per-node dense layers (tiny: ~0.54 GFLOP total), H written transposed bf16

__global__ __launch_bounds__(256) void dense_kernel(
    const float* __restrict__ X1, const float* __restrict__ X2,
    const float* __restrict__ X3, int inDim,
    const float* __restrict__ w1, const float* __restrict__ b1,
    const float* __restrict__ w2, const float* __restrict__ b2,
    const float* __restrict__ w3, const float* __restrict__ b3,
    __bf16* __restrict__ Htcl, __bf16* __restrict__ Htue,
    int computeUe, int n) {
  const int idx = blockIdx.x * blockDim.x + threadIdx.x;
  const int row = idx >> 6;
  const int col = idx & 63;
  if (row >= n) return;

  const float* x1 = X1 + (size_t)row * inDim;
  const float* x2 = X2 + (size_t)row * inDim;
  float s1 = b1[col], s2 = b2[col];
  for (int k = 0; k < inDim; ++k) {
    s1 += x1[k] * w1[k * 64 + col];
    s2 += x2[k] * w2[k * 64 + col];
  }
  const float h = fmaxf(s1, 0.f) + fmaxf(s2, 0.f);
  Htcl[(size_t)col * n + row] = f2bf(h);

  if (computeUe) {
    const float* x3 = X3 + (size_t)row * inDim;
    float s3 = b3[col];
    for (int k = 0; k < inDim; ++k) s3 += x3[k] * w3[k * 64 + col];
    Htue[(size_t)col * n + row] = f2bf(fmaxf(s3, 0.f));
  }
}

// --- column-sum pooling of final H_cl + 2-layer MLP head -> scalar ---------

__global__ __launch_bounds__(256) void pool_head_kernel(
    const __bf16* __restrict__ Htcl,   // [64 x n]
    const float* __restrict__ Qw1, const float* __restrict__ Qb1,
    const float* __restrict__ Qw2, const float* __restrict__ Qb2,
    float* __restrict__ out, int n) {
  __shared__ float red[256];
  __shared__ float pooled[64];
  __shared__ float h[64];
  const int tid  = threadIdx.x;
  const int col  = tid >> 2;           // 0..63
  const int part = tid & 3;            // 4-way split over rows

  float s = 0.f;
  const __bf16* cp = Htcl + (size_t)col * n;
  for (int r = part; r < n; r += 4) s += bf2f(cp[r]);
  red[tid] = s;
  __syncthreads();

  if (part == 0)
    pooled[col] = red[tid] + red[tid + 1] + red[tid + 2] + red[tid + 3];
  __syncthreads();

  if (tid < 64) {
    float acc = Qb1[tid];
    for (int k = 0; k < 64; ++k) acc += pooled[k] * Qw1[k * 64 + tid];
    h[tid] = fmaxf(acc, 0.f);
  }
  __syncthreads();

  if (tid == 0) {
    float acc = Qb2[0];
    for (int j = 0; j < 64; ++j) acc += h[j] * Qw2[j];
    out[0] = acc;
  }
}

// ---------------------------------------------------------------------------

extern "C" void kernel_launch(void* const* d_in, const int* in_sizes, int n_in,
                              void* d_out, int out_size, void* d_ws, size_t ws_size,
                              hipStream_t stream) {
  constexpr int N = 8192, D = 64;
  const float* Xcl1_in = (const float*)d_in[0];
  const float* Xcl2_in = (const float*)d_in[1];
  const float* Xue_in  = (const float*)d_in[2];
  const float* A_cl    = (const float*)d_in[3];
  const float* A_ue    = (const float*)d_in[4];
  const float* W1w0 = (const float*)d_in[5];  const float* W1b0 = (const float*)d_in[6];
  const float* W1w  = (const float*)d_in[7];  const float* W1b  = (const float*)d_in[8];
  const float* W2w0 = (const float*)d_in[9];  const float* W2b0 = (const float*)d_in[10];
  const float* W2w  = (const float*)d_in[11]; const float* W2b  = (const float*)d_in[12];
  const float* W3w0 = (const float*)d_in[13]; const float* W3b0 = (const float*)d_in[14];
  const float* W3w  = (const float*)d_in[15]; const float* W3b  = (const float*)d_in[16];
  const float* Qw1  = (const float*)d_in[17]; const float* Qb1  = (const float*)d_in[18];
  const float* Qw2  = (const float*)d_in[19]; const float* Qb2  = (const float*)d_in[20];

  // workspace layout:
  //   3 fp32 X buffers [N x 64]  (6 MB)
  //   2 bf16 Ht buffers [64 x N] (2 MB)
  //   optionally 2 bf16 A copies [N x N] (256 MB) if ws_size allows
  float*  Xc1  = (float*)d_ws;
  float*  Xc2  = Xc1 + (size_t)N * D;
  float*  Xue  = Xc2 + (size_t)N * D;
  __bf16* Htcl = (__bf16*)(Xue + (size_t)N * D);
  __bf16* Htue = Htcl + (size_t)N * D;
  __bf16* Abcl = Htue + (size_t)N * D;
  __bf16* Abue = Abcl + (size_t)N * N;

  const size_t needBig = (size_t)N * D * (3 * 4 + 2 * 2) + (size_t)N * N * 2 * 2;
  const bool   bigws   = ws_size >= needBig;

  const dim3 dgrid((N * D) / 256), dblk(256);
  const dim3 ggrid(N / 64), gblk(128);

  if (bigws) {
    // one-shot fp32 -> bf16 conversion of both adjacency matrices;
    // each bf16 A is 128 MB and can stay resident in the 192 MB L2.
    const size_t cnt8 = (size_t)N * N / 8;
    convert_kernel<<<dim3(cnt8 / 256), dblk, 0, stream>>>(A_cl, Abcl, cnt8);
    convert_kernel<<<dim3(cnt8 / 256), dblk, 0, stream>>>(A_ue, Abue, cnt8);
  }

  // layer 0 (inDim = 2, separate weights)
  dense_kernel<<<dgrid, dblk, 0, stream>>>(Xcl1_in, Xcl2_in, Xue_in, 2,
      W1w0, W1b0, W2w0, W2b0, W3w0, W3b0, Htcl, Htue, 1, N);

  for (int l = 0; l <= 2; ++l) {
    if (l > 0) {
      dense_kernel<<<dgrid, dblk, 0, stream>>>(Xc1, Xc2, Xue, D,
          W1w + (l - 1) * D * D, W1b + (l - 1) * D,
          W2w + (l - 1) * D * D, W2b + (l - 1) * D,
          W3w + (l - 1) * D * D, W3b + (l - 1) * D,
          Htcl, Htue, 1, N);
    }
    if (bigws) {
      gemm_kernel<1, 1><<<ggrid, gblk, 0, stream>>>(Abue, Htue, Htcl, Xc2, Xue, N);
      gemm_kernel<0, 1><<<ggrid, gblk, 0, stream>>>(Abcl, Htcl, nullptr, Xc1, nullptr, N);
    } else {
      gemm_kernel<1, 0><<<ggrid, gblk, 0, stream>>>(A_ue, Htue, Htcl, Xc2, Xue, N);
      gemm_kernel<0, 0><<<ggrid, gblk, 0, stream>>>(A_cl, Htcl, nullptr, Xc1, nullptr, N);
    }
  }

  // layer 3: only H_cl is live afterwards (A-matmuls and H_ue are dead code)
  dense_kernel<<<dgrid, dblk, 0, stream>>>(Xc1, Xc2, Xue, D,
      W1w + 2 * D * D, W1b + 2 * D,
      W2w + 2 * D * D, W2b + 2 * D,
      W3w + 2 * D * D, W3b + 2 * D,
      Htcl, nullptr, 0, N);

  pool_head_kernel<<<1, 256, 0, stream>>>(Htcl, Qw1, Qb1, Qw2, Qb2,
                                          (float*)d_out, N);
}
